// StateMixer_54107997995556
// MI455X (gfx1250) — matrix operand order repaired and checked by reading, compiled, not run on hardware
//
#include <hip/hip_runtime.h>
#include <hip/hip_bf16.h>
#include <math.h>

typedef __attribute__((ext_vector_type(16))) __bf16 v16bf;
typedef __attribute__((ext_vector_type(8)))  float  v8f;
typedef __attribute__((ext_vector_type(8)))  __bf16 bf16x8;
typedef __attribute__((ext_vector_type(4)))  __bf16 bf16x4;

#define CC 128

static __device__ __forceinline__ __bf16 f2bf(float f) {
  // round-to-nearest-even fp32 -> bf16
  unsigned u = __builtin_bit_cast(unsigned, f);
  unsigned r = (u + 0x7FFFu + ((u >> 16) & 1u)) >> 16;
  unsigned short h = (unsigned short)r;
  return __builtin_bit_cast(__bf16, h);
}

// ---------------------------------------------------------------------------
// Kernel 0: per node type, convert Wl -> bf16 and compute d = Wr @ tok + br
// ---------------------------------------------------------------------------
__global__ __launch_bounds__(128) void prep_kernel(
    const float* __restrict__ Wl, const float* __restrict__ Wr,
    const float* __restrict__ br, const float* __restrict__ tok,
    __bf16* __restrict__ wlbf, float* __restrict__ dvec)
{
  const int n = threadIdx.x;               // 128 threads, 1 block
  float a = br[n];
  #pragma unroll 4
  for (int k = 0; k < CC; ++k) a += Wr[n * CC + k] * tok[k];
  dvec[n] = a;
  #pragma unroll 4
  for (int k = 0; k < CC; ++k) wlbf[n * CC + k] = f2bf(Wl[n * CC + k]);
}

// ---------------------------------------------------------------------------
// Kernel 1: GAT pooling main pass.  One launch per node type.
// Block = 256 threads = 8 wave32. Each wave handles 16-row tiles, computing
// S[16,128] = X_tile @ Wl^T with v_wmma_f32_16x16x32_bf16, logits via butterfly
// shuffles, and an online-softmax weighted accumulation of S.
// ---------------------------------------------------------------------------
__global__ __launch_bounds__(256) void gat_kernel(
    const float* __restrict__ x, int nrows,
    const __bf16* __restrict__ wlbf,
    const float* __restrict__ bl,
    const float* __restrict__ dvec,
    const float* __restrict__ att,
    float* __restrict__ pmax, float* __restrict__ psum,
    float* __restrict__ pvec)
{
  __shared__ __align__(32) __bf16 wlds[CC * CC];        // 32 KB  Wl (bf16)
  __shared__ __align__(32) __bf16 xlds[8][16 * CC];     // 32 KB  per-wave X tile
  __shared__ float rmax[16], rsum[16];
  __shared__ float racc[16][CC];                        // 8 KB   half-wave partials

  const int tid  = threadIdx.x;
  const int lane = tid & 31;
  const int wid  = tid >> 5;
  const int ln   = lane & 15;
  const int hi   = lane >> 4;

  // Stage Wl.bf16 into LDS (coalesced b128)
  {
    const uint4* src = (const uint4*)wlbf;
    uint4*       dst = (uint4*)wlds;
    for (int i = tid; i < (CC * CC * 2) / 16; i += 256) dst[i] = src[i];
  }
  __syncthreads();

  // Per-lane constants: this lane touches columns n = nt*16 + ln
  float pre_bl[8], pre_d[8], pre_att[8];
  #pragma unroll
  for (int nt = 0; nt < 8; ++nt) {
    const int n = nt * 16 + ln;
    pre_bl[nt]  = bl[n];
    pre_d[nt]   = dvec[n];
    pre_att[nt] = att[n];
  }

  float run_max = -INFINITY, run_sum = 0.f;
  float acc[8];
  #pragma unroll
  for (int nt = 0; nt < 8; ++nt) acc[nt] = 0.f;

  const int ntiles = nrows >> 4;
  for (int tile = blockIdx.x * 8 + wid; tile < ntiles; tile += gridDim.x * 8) {
    const int r0 = tile << 4;

    // Load 16x128 fp32 rows (512B/row coalesced), convert to bf16, stage to LDS
    #pragma unroll
    for (int r = 0; r < 16; ++r) {
      const float4 v = ((const float4*)(x + (size_t)(r0 + r) * CC))[lane];
      bf16x4 p;
      p[0] = f2bf(v.x); p[1] = f2bf(v.y); p[2] = f2bf(v.z); p[3] = f2bf(v.w);
      *(bf16x4*)&xlds[wid][r * CC + lane * 4] = p;
    }
    // DS ops are in-order within a wave; fence so the cross-lane reads below
    // are not reordered above the stores.
    asm volatile("s_wait_dscnt 0" ::: "memory");

    // A fragments: 16-bit A 16x32; lanes 0-15: K0-15, lanes 16-31: K16-31
    v16bf a[4];
    #pragma unroll
    for (int kc = 0; kc < 4; ++kc) {
      const bf16x8 lo = *(const bf16x8*)&xlds[wid][ln * CC + kc * 32 + hi * 16];
      const bf16x8 hv = *(const bf16x8*)&xlds[wid][ln * CC + kc * 32 + hi * 16 + 8];
      v16bf av;
      #pragma unroll
      for (int i = 0; i < 8; ++i) { av[i] = lo[i]; av[i + 8] = hv[i]; }
      a[kc] = av;
    }

    // S = A @ Wl^T : 8 column tiles x 4 K chunks of WMMA
    v8f cf[8];
    #pragma unroll
    for (int nt = 0; nt < 8; ++nt) {
      v8f c = {};
      #pragma unroll
      for (int kc = 0; kc < 4; ++kc) {
        // B[k,n] = Wl[n,k]; lane = column n, half-wave selects K half (contiguous 32B)
        const v16bf b = *(const v16bf*)&wlds[(nt * 16 + ln) * CC + kc * 32 + hi * 16];
        c = __builtin_amdgcn_wmma_f32_16x16x32_bf16(false, a[kc], false, b,
                                                    (short)0, c, false, false);
      }
      cf[nt] = c;
    }

    // logits: leaky_relu(s + d) . att   (s includes bl)
    float lp[8];
    #pragma unroll
    for (int r = 0; r < 8; ++r) lp[r] = 0.f;
    #pragma unroll
    for (int nt = 0; nt < 8; ++nt) {
      #pragma unroll
      for (int r = 0; r < 8; ++r) {
        float s = cf[nt][r] + pre_bl[nt];
        cf[nt][r] = s;                          // keep s for weighted sum
        float h = s + pre_d[nt];
        h = (h > 0.f) ? h : 0.2f * h;
        lp[r] += h * pre_att[nt];
      }
    }
    #pragma unroll
    for (int r = 0; r < 8; ++r) {               // reduce over 16 columns-lanes
      float v = lp[r];
      v += __shfl_xor(v, 1, 32);
      v += __shfl_xor(v, 2, 32);
      v += __shfl_xor(v, 4, 32);
      v += __shfl_xor(v, 8, 32);
      lp[r] = v;                                 // uniform within each half-wave
    }

    // online softmax accumulate (per half-wave: rows r (+8 for hi half))
    float tmax = lp[0];
    #pragma unroll
    for (int r = 1; r < 8; ++r) tmax = fmaxf(tmax, lp[r]);
    const float nm = fmaxf(run_max, tmax);
    const float sc = __expf(run_max - nm);       // exp(-inf - finite) = 0 first pass
    run_sum *= sc;
    #pragma unroll
    for (int nt = 0; nt < 8; ++nt) acc[nt] *= sc;
    float w[8];
    #pragma unroll
    for (int r = 0; r < 8; ++r) { w[r] = __expf(lp[r] - nm); run_sum += w[r]; }
    #pragma unroll
    for (int nt = 0; nt < 8; ++nt) {
      float av = acc[nt];
      #pragma unroll
      for (int r = 0; r < 8; ++r) av += w[r] * cf[nt][r];
      acc[nt] = av;
    }
    run_max = nm;
  }

  // Block combine: 16 half-wave partials -> one (max, sum, vec128) per block
  const int pidx = wid * 2 + hi;
  #pragma unroll
  for (int nt = 0; nt < 8; ++nt) racc[pidx][nt * 16 + ln] = acc[nt];
  if (ln == 0) { rmax[pidx] = run_max; rsum[pidx] = run_sum; }
  __syncthreads();

  if (tid < CC) {
    const int n = tid;
    float gm = -INFINITY;
    #pragma unroll
    for (int p = 0; p < 16; ++p) gm = fmaxf(gm, rmax[p]);
    if (gm == -INFINITY) {                       // block got zero tiles
      pvec[(size_t)blockIdx.x * CC + n] = 0.f;
      if (n == 0) { pmax[blockIdx.x] = -INFINITY; psum[blockIdx.x] = 0.f; }
    } else {
      float tot = 0.f, v = 0.f;
      #pragma unroll
      for (int p = 0; p < 16; ++p) {
        const float f = __expf(rmax[p] - gm);
        tot += f * rsum[p];
        v   += f * racc[p][n];
      }
      pvec[(size_t)blockIdx.x * CC + n] = v;
      if (n == 0) { pmax[blockIdx.x] = gm; psum[blockIdx.x] = tot; }
    }
  }
}

// ---------------------------------------------------------------------------
// Kernel 2: merge block partials, softmax-normalize, + bias, layernorm, tanh
// ---------------------------------------------------------------------------
static __device__ __forceinline__ float blocksum128(float v, float* red) {
  v += __shfl_xor(v, 1, 32);
  v += __shfl_xor(v, 2, 32);
  v += __shfl_xor(v, 4, 32);
  v += __shfl_xor(v, 8, 32);
  v += __shfl_xor(v, 16, 32);
  const int j = threadIdx.x;
  if ((j & 31) == 0) red[j >> 5] = v;
  __syncthreads();
  const float s = red[0] + red[1] + red[2] + red[3];
  __syncthreads();
  return s;
}

__global__ __launch_bounds__(128) void gat_finalize_kernel(
    const float* __restrict__ pmax, const float* __restrict__ psum,
    const float* __restrict__ pvec, int nblk,
    const float* __restrict__ bias, const float* __restrict__ lng,
    const float* __restrict__ lnb,
    float* __restrict__ feat_out, float* __restrict__ feat_stage)
{
  __shared__ float red[4];
  const int n = threadIdx.x;
  float gm = -INFINITY;
  for (int b = 0; b < nblk; ++b) gm = fmaxf(gm, pmax[b]);
  float tot = 0.f, v = 0.f;
  for (int b = 0; b < nblk; ++b) {
    const float f = __expf(pmax[b] - gm);
    tot += f * psum[b];
    v   += f * pvec[(size_t)b * CC + n];
  }
  const float o = v / tot + bias[n];
  const float mean = blocksum128(o, red) * (1.f / 128.f);
  const float d = o - mean;
  const float var = blocksum128(d * d, red) * (1.f / 128.f);
  const float feat = tanhf(d * rsqrtf(var + 1e-5f) * lng[n] + lnb[n]);
  feat_out[n] = feat;
  feat_stage[n] = feat;
}

// ---------------------------------------------------------------------------
// Kernel 3: graph_mix MLP on z[448] -> graph_feature[256] (tiny; single block)
// ---------------------------------------------------------------------------
static __device__ __forceinline__ float blocksum512(float v, float* red) {
  v += __shfl_xor(v, 1, 32);
  v += __shfl_xor(v, 2, 32);
  v += __shfl_xor(v, 4, 32);
  v += __shfl_xor(v, 8, 32);
  v += __shfl_xor(v, 16, 32);
  const int j = threadIdx.x;
  if ((j & 31) == 0) red[j >> 5] = v;
  __syncthreads();
  float s = 0.f;
  #pragma unroll
  for (int i = 0; i < 16; ++i) s += red[i];
  __syncthreads();
  return s;
}

__global__ __launch_bounds__(512) void mix_kernel(
    const float* __restrict__ gattr, const float* __restrict__ featbuf, // 384
    const float* __restrict__ w1,  const float* __restrict__ b1,
    const float* __restrict__ w2,  const float* __restrict__ b2,
    const float* __restrict__ pw,  const float* __restrict__ pb,
    const float* __restrict__ lg1, const float* __restrict__ lb1,
    const float* __restrict__ rw1, const float* __restrict__ rb1,
    const float* __restrict__ rw2, const float* __restrict__ rb2,
    const float* __restrict__ lg2, const float* __restrict__ lb2,
    const float* __restrict__ fw,  const float* __restrict__ fb,
    float* __restrict__ out)
{
  __shared__ float z[448];
  __shared__ float buf[512];
  __shared__ float h[512];
  __shared__ float red[16];
  const int j = threadIdx.x;
  if (j < 64) z[j] = gattr[j];
  else if (j < 448) z[j] = featbuf[j - 64];
  __syncthreads();

  // rl1: t1 = tanh(z@w1^T+b1); proj = z@pw^T+pb
  float t1 = b1[j], pj = pb[j];
  #pragma unroll 4
  for (int k = 0; k < 448; ++k) {
    t1 += w1[j * 448 + k] * z[k];
    pj += pw[j * 448 + k] * z[k];
  }
  buf[j] = tanhf(t1);
  __syncthreads();
  float r = pj + b2[j];
  #pragma unroll 4
  for (int k = 0; k < 512; ++k) r += w2[j * 512 + k] * buf[k];
  float mean = blocksum512(r, red) * (1.f / 512.f);
  float d = r - mean;
  float var = blocksum512(d * d, red) * (1.f / 512.f);
  h[j] = tanhf(d * rsqrtf(var + 1e-5f) * lg1[j] + lb1[j]);
  __syncthreads();

  // rl2 (identity projection)
  float t2 = rb1[j];
  #pragma unroll 4
  for (int k = 0; k < 512; ++k) t2 += rw1[j * 512 + k] * h[k];
  buf[j] = tanhf(t2);
  __syncthreads();
  float r2 = h[j] + rb2[j];
  #pragma unroll 4
  for (int k = 0; k < 512; ++k) r2 += rw2[j * 512 + k] * buf[k];
  mean = blocksum512(r2, red) * (1.f / 512.f);
  d = r2 - mean;
  var = blocksum512(d * d, red) * (1.f / 512.f);
  h[j] = tanhf(d * rsqrtf(var + 1e-5f) * lg2[j] + lb2[j]);
  __syncthreads();

  if (j < 256) {
    float o = fb[j];
    #pragma unroll 4
    for (int k = 0; k < 512; ++k) o += fw[j * 512 + k] * h[k];
    out[j] = o;
  }
}

// ---------------------------------------------------------------------------
// Launcher
// ---------------------------------------------------------------------------
extern "C" void kernel_launch(void* const* d_in, const int* in_sizes, int n_in,
                              void* d_out, int out_size, void* d_ws, size_t ws_size,
                              hipStream_t stream)
{
  // Input layout (setup_inputs dict order, None leaves dropped):
  // 0 x_operation, 1 x_machine, 2 x_AGV, 3 global_attr,
  // 4+9t .. : per-type {Wl, bl, Wr, br, att, bias, ln_g, ln_b, tok}
  // 31..46  : mix {rl1: w1,b1,w2,b2,pw,pb,g,be; rl2: w1,b1,w2,b2,g,be; fw,fb}
  const float* x[3]  = {(const float*)d_in[0], (const float*)d_in[1], (const float*)d_in[2]};
  const int nrows[3] = {in_sizes[0] / CC, in_sizes[1] / CC, in_sizes[2] / CC};
  const float* gattr = (const float*)d_in[3];
  auto P = [&](int t, int i) { return (const float*)d_in[4 + 9 * t + i]; };
  auto M = [&](int i) { return (const float*)d_in[31 + i]; };

  // Workspace carve-up (shrink block count if ws is small)
  auto rnd = [](size_t b) { return (b + 255) & ~size_t(255); };
  int nblk = 512;
  while (nblk > 8) {
    const size_t need = 3 * (rnd(CC * CC * 2) + rnd(CC * 4) + 2 * rnd((size_t)nblk * 4) +
                             rnd((size_t)nblk * CC * 4)) + rnd(384 * 4);
    if (need <= ws_size) break;
    nblk >>= 1;
  }
  char* ws = (char*)d_ws;
  size_t off = 0;
  auto alloc = [&](size_t bytes) { void* p = ws + off; off += rnd(bytes); return p; };
  __bf16* wlbf[3]; float *dvec[3], *pm[3], *ps[3], *pv[3];
  for (int t = 0; t < 3; ++t) {
    wlbf[t] = (__bf16*)alloc(CC * CC * 2);
    dvec[t] = (float*)alloc(CC * 4);
    pm[t]   = (float*)alloc((size_t)nblk * 4);
    ps[t]   = (float*)alloc((size_t)nblk * 4);
    pv[t]   = (float*)alloc((size_t)nblk * CC * 4);
  }
  float* featbuf = (float*)alloc(384 * 4);

  for (int t = 0; t < 3; ++t)
    prep_kernel<<<1, 128, 0, stream>>>(P(t, 0), P(t, 2), P(t, 3), P(t, 8),
                                       wlbf[t], dvec[t]);
  for (int t = 0; t < 3; ++t)
    gat_kernel<<<nblk, 256, 0, stream>>>(x[t], nrows[t], wlbf[t],
                                         P(t, 1), dvec[t], P(t, 4),
                                         pm[t], ps[t], pv[t]);
  float* outp = (float*)d_out;
  for (int t = 0; t < 3; ++t)
    gat_finalize_kernel<<<1, 128, 0, stream>>>(pm[t], ps[t], pv[t], nblk,
                                               P(t, 5), P(t, 6), P(t, 7),
                                               outp + t * CC, featbuf + t * CC);
  mix_kernel<<<1, 512, 0, stream>>>(gattr, featbuf,
                                    M(0), M(1), M(2), M(3), M(4), M(5), M(6), M(7),
                                    M(8), M(9), M(10), M(11), M(12), M(13),
                                    M(14), M(15),
                                    outp + 384);
}
